// MSA_50457275793597
// MI455X (gfx1250) — compile-verified, matrix operand order checked by reading, not run on hardware
//
#include <hip/hip_runtime.h>
#include <hip/hip_bf16.h>

// ---- problem constants -----------------------------------------------------
#define B_  8
#define S_  1024
#define DM_ 768
#define H_  12
#define HD_ 64
#define BS_ (B_ * S_)          // 8192 token rows

typedef __attribute__((ext_vector_type(16))) _Float16 v16h;
typedef __attribute__((ext_vector_type(8)))  _Float16 v8h;
typedef __attribute__((ext_vector_type(8)))  float    v8f;

// ---- WMMA helpers ----------------------------------------------------------
__device__ __forceinline__ v8f wmma_f16(v16h a, v16h b, v8f c) {
  // D = A(16x32 f16) x B(32x16 f16) + C(16x16 f32)
  return __builtin_amdgcn_wmma_f32_16x16x32_f16(
      /*neg_a=*/false, a, /*neg_b=*/false, b,
      /*c_mod=*/(short)0, c, /*reuse_a=*/false, /*reuse_b=*/false);
}

// Row-major 16x32 fragment (A layout; also used for B by feeding B-transposed,
// which is column-major-by-lane == same layout on the N index).
// ISA 7.12.2: lanes 0-15 hold M=lane, K={0..7,16..23}; lanes 16-31: K={8..15,24..31}.
__device__ __forceinline__ v16h load_frag_row(const _Float16* tile, int ld) {
  const int lane = threadIdx.x & 31;
  const int r    = lane & 15;
  const int kb   = (lane & 16) ? 8 : 0;
  const _Float16* p = tile + (size_t)r * ld + kb;
  v8h lo = *reinterpret_cast<const v8h*>(p);        // 16B load
  v8h hi = *reinterpret_cast<const v8h*>(p + 16);   // 16B load
  v16h f;
#pragma unroll
  for (int i = 0; i < 8; ++i) { f[i] = lo[i]; f[i + 8] = hi[i]; }
  return f;
}

// Transposed gather from an LDS tile stored [32 rows(k)][64 cols].
__device__ __forceinline__ v16h load_frag_trans_lds(const _Float16* lds, int col0) {
  const int lane = threadIdx.x & 31;
  const int m    = lane & 15;
  const int kb   = (lane & 16) ? 8 : 0;
  v16h f;
#pragma unroll
  for (int i = 0; i < 8; ++i) {
    f[i]     = lds[(kb + i)      * 64 + col0 + m];
    f[i + 8] = lds[(kb + 16 + i) * 64 + col0 + m];
  }
  return f;
}

// CDNA5 direct global->LDS async DMA (16B per lane), tracked by ASYNCcnt.
// LDS operand = byte offset within the workgroup's LDS allocation; for a
// generic pointer to __shared__ the low 32 bits are exactly that offset.
__device__ __forceinline__ void async_copy_b128(_Float16* lds_ptr,
                                                const _Float16* gptr) {
  const unsigned ldsoff = (unsigned)(uintptr_t)lds_ptr;
  asm volatile("global_load_async_to_lds_b128 %0, %1, off"
               :: "v"(ldsoff), "v"(gptr)
               : "memory");
}
__device__ __forceinline__ void wait_async0() {
  asm volatile("s_wait_asynccnt 0" ::: "memory");
}

// ---- kernels ---------------------------------------------------------------

// fp32 -> fp16 convert, 8 elements per thread (b128 in/out)
__global__ void k_cvt_f16v(const float* __restrict__ in,
                           _Float16* __restrict__ out, int n8) {
  int i = blockIdx.x * blockDim.x + threadIdx.x;
  if (i < n8) {
    const v8f x = *reinterpret_cast<const v8f*>(in + 8 * (size_t)i);
    v8h o;
#pragma unroll
    for (int j = 0; j < 8; ++j) o[j] = (_Float16)x[j];
    *reinterpret_cast<v8h*>(out + 8 * (size_t)i) = o;
  }
}

// Y[M=8192, N=768] = Xh @ Wh^T + b, optionally scaled; OutT in {f16, f32}.
// One wave computes a 16x64 strip (4 N-tiles). 768 blocks x 256 threads.
template <typename OutT>
__global__ __launch_bounds__(256) void k_gemm_xwT(
    const _Float16* __restrict__ A,   // [8192, 768] row-major
    const _Float16* __restrict__ W,   // [768, 768] row-major (Linear weight)
    const float* __restrict__ bias,   // [768]
    OutT* __restrict__ out,           // [8192, 768]
    float scale) {
  const int waveId = blockIdx.x * 8 + (threadIdx.x >> 5);
  const int rb = waveId / (DM_ / 64);   // 512 row blocks of 16
  const int cb = waveId % (DM_ / 64);   // 12 col blocks of 64
  const _Float16* abase = A + (size_t)rb * 16 * DM_;
  const _Float16* wbase = W + (size_t)cb * 64 * DM_;

  v8f acc[4] = {v8f{}, v8f{}, v8f{}, v8f{}};
  for (int k0 = 0; k0 < DM_; k0 += 32) {
    v16h af = load_frag_row(abase + k0, DM_);
#pragma unroll
    for (int nt = 0; nt < 4; ++nt) {
      v16h bf = load_frag_row(wbase + (size_t)nt * 16 * DM_ + k0, DM_);
      acc[nt] = wmma_f16(af, bf, acc[nt]);
    }
  }
  const int lane = threadIdx.x & 31;
  const int nl = lane & 15;
  const int ma = (lane & 16) ? 8 : 0;
#pragma unroll
  for (int nt = 0; nt < 4; ++nt) {
    const int n = cb * 64 + nt * 16 + nl;
    const float bv = bias[n];
#pragma unroll
    for (int j = 0; j < 8; ++j) {
      const int m = rb * 16 + j + ma;
      out[(size_t)m * DM_ + n] = (OutT)((acc[nt][j] + bv) * scale);
    }
  }
}

// Fused energy + softmax:
//   attention[b,h,s1,:] = softmax( q[b,h,s1,:]/8 . k[b,h,:,:]^T )
// One wave owns 16 rows. Pass 1 streams q.k^T tiles through WMMA keeping only
// online (rowmax, sum-exp) stats; pass 2 recomputes tiles and writes softmax
// with non-temporal stores. 2x WMMA flops buys removal of 800 MB of HBM
// round-trip -- a clear win at 23.3 TB/s.
__global__ __launch_bounds__(256) void k_energy_softmax(
    const _Float16* __restrict__ qh, const _Float16* __restrict__ kh,
    float* __restrict__ attn) {
  const int waveId = blockIdx.x * 8 + (threadIdx.x >> 5);
  const int sb = waveId & 63;       // S/16 = 64
  const int bh = waveId >> 6;       // 0..95
  const int b = bh / H_, h = bh % H_;
  const _Float16* qbase = qh + ((size_t)b * S_ + sb * 16) * DM_ + h * HD_;
  const _Float16* kbase = kh + (size_t)b * S_ * DM_ + h * HD_;
  float* orow = attn + ((size_t)bh * S_ + sb * 16) * S_;

  const v16h a0 = load_frag_row(qbase, DM_);        // K = 0..31
  const v16h a1 = load_frag_row(qbase + 32, DM_);   // K = 32..63
  const int lane = threadIdx.x & 31;
  const int nl = lane & 15;
  const int ma = (lane & 16) ? 8 : 0;

  float m[8], ssum[8];
#pragma unroll
  for (int j = 0; j < 8; ++j) { m[j] = -3.402823e38f; ssum[j] = 0.f; }

  // ---- pass 1: online row stats -------------------------------------------
  for (int n0 = 0; n0 < S_; n0 += 16) {
    v16h b0 = load_frag_row(kbase + (size_t)n0 * DM_, DM_);
    v16h b1 = load_frag_row(kbase + (size_t)n0 * DM_ + 32, DM_);
    v8f c = v8f{};
    c = wmma_f16(a0, b0, c);
    c = wmma_f16(a1, b1, c);
#pragma unroll
    for (int j = 0; j < 8; ++j) {
      const float x = c[j];
      const float mn = fmaxf(m[j], x);
      ssum[j] = ssum[j] * __expf(m[j] - mn) + __expf(x - mn);
      m[j] = mn;
    }
  }
  // combine partial (max, sum) across the 16 lanes of each half-wave
#pragma unroll
  for (int j = 0; j < 8; ++j) {
#pragma unroll
    for (int off = 8; off > 0; off >>= 1) {
      const float mo = __shfl_xor(m[j], off, 32);
      const float so = __shfl_xor(ssum[j], off, 32);
      const float mn = fmaxf(m[j], mo);
      ssum[j] = ssum[j] * __expf(m[j] - mn) + so * __expf(mo - mn);
      m[j] = mn;
    }
    ssum[j] = 1.0f / ssum[j];
  }
  // ---- pass 2: recompute + write softmax (streaming, non-temporal) --------
  for (int n0 = 0; n0 < S_; n0 += 16) {
    v16h b0 = load_frag_row(kbase + (size_t)n0 * DM_, DM_);
    v16h b1 = load_frag_row(kbase + (size_t)n0 * DM_ + 32, DM_);
    v8f c = v8f{};
    c = wmma_f16(a0, b0, c);
    c = wmma_f16(a1, b1, c);
#pragma unroll
    for (int j = 0; j < 8; ++j)
      __builtin_nontemporal_store(__expf(c[j] - m[j]) * ssum[j],
                                  &orow[(size_t)(j + ma) * S_ + n0 + nl]);
  }
}

// Mt[bh][d2][d1] = sum_s kh[b,s,hd+d1] * vh[b,s,hd+d2]  (stored transposed)
// One workgroup (4 waves) per (b,h); k/v tiles staged into LDS via the CDNA5
// async global->LDS DMA path (ASYNCcnt), then consumed by transposed gathers.
__global__ __launch_bounds__(128) void k_ktv(
    const _Float16* __restrict__ kh, const _Float16* __restrict__ vh,
    _Float16* __restrict__ Mt) {
  const int bh = blockIdx.x;
  const int b = bh / H_, h = bh % H_;
  const _Float16* kbase = kh + (size_t)b * S_ * DM_ + h * HD_;
  const _Float16* vbase = vh + (size_t)b * S_ * DM_ + h * HD_;
  __shared__ _Float16 lk[32 * 64];
  __shared__ _Float16 lv[32 * 64];
  const int wave = threadIdx.x >> 5;
  const int lane = threadIdx.x & 31;

  v8f acc[4] = {v8f{}, v8f{}, v8f{}, v8f{}};
  for (int s0 = 0; s0 < S_; s0 += 32) {
    __syncthreads();   // previous tile fully consumed before overwrite
    // 32x64 f16 = 256 16B-chunks per tile; 128 threads -> 2 chunks each
    for (int c = threadIdx.x; c < 256; c += 128) {
      const int row = c >> 3, col = (c & 7) * 8;
      async_copy_b128(&lk[row * 64 + col],
                      &kbase[(size_t)(s0 + row) * DM_ + col]);
      async_copy_b128(&lv[row * 64 + col],
                      &vbase[(size_t)(s0 + row) * DM_ + col]);
    }
    wait_async0();
    __syncthreads();
    // A = k^T (16 d1-rows owned by this wave), B^T = v^T (transposed gather)
    v16h af = load_frag_trans_lds(lk, wave * 16);
#pragma unroll
    for (int nt = 0; nt < 4; ++nt) {
      v16h bf = load_frag_trans_lds(lv, nt * 16);
      acc[nt] = wmma_f16(af, bf, acc[nt]);
    }
  }
  const int nl = lane & 15;
  const int ma = (lane & 16) ? 8 : 0;
  _Float16* mbase = Mt + (size_t)bh * (HD_ * HD_);
#pragma unroll
  for (int nt = 0; nt < 4; ++nt) {
    const int d2 = nt * 16 + nl;
#pragma unroll
    for (int j = 0; j < 8; ++j) {
      const int d1 = wave * 16 + j + ma;
      mbase[(size_t)d2 * HD_ + d1] = (_Float16)acc[nt][j];  // transposed store
    }
  }
}

// oh[b,s,hd+d2] = sum_d1 qh[b,s,hd+d1] * M[d1,d2]  (qh pre-scaled by 1/8)
// Mt is [d2][d1] so B^T reads row-major. One wave: 16 s-rows x 64 cols.
__global__ __launch_bounds__(256) void k_qm(
    const _Float16* __restrict__ qh, const _Float16* __restrict__ Mt,
    _Float16* __restrict__ oh) {
  const int waveId = blockIdx.x * 8 + (threadIdx.x >> 5);
  const int sb = waveId & 63;
  const int bh = waveId >> 6;
  const int b = bh / H_, h = bh % H_;
  const _Float16* qbase = qh + ((size_t)b * S_ + sb * 16) * DM_ + h * HD_;
  const _Float16* mt = Mt + (size_t)bh * (HD_ * HD_);
  _Float16* obase = oh + ((size_t)b * S_ + sb * 16) * DM_ + h * HD_;

  const v16h a0 = load_frag_row(qbase, DM_);
  const v16h a1 = load_frag_row(qbase + 32, DM_);
  const int lane = threadIdx.x & 31;
  const int nl = lane & 15;
  const int ma = (lane & 16) ? 8 : 0;
#pragma unroll
  for (int nt = 0; nt < 4; ++nt) {
    v16h b0 = load_frag_row(mt + (size_t)nt * 16 * HD_, HD_);
    v16h b1 = load_frag_row(mt + (size_t)nt * 16 * HD_ + 32, HD_);
    v8f c = v8f{};
    c = wmma_f16(a0, b0, c);
    c = wmma_f16(a1, b1, c);
#pragma unroll
    for (int j = 0; j < 8; ++j)
      obase[(size_t)(j + ma) * DM_ + nt * 16 + nl] = (_Float16)c[j];
  }
}

// ---- launch ----------------------------------------------------------------
extern "C" void kernel_launch(void* const* d_in, const int* in_sizes, int n_in,
                              void* d_out, int out_size, void* d_ws, size_t ws_size,
                              hipStream_t stream) {
  const float* x  = (const float*)d_in[0];
  const float* Wq = (const float*)d_in[1];
  const float* bq = (const float*)d_in[2];
  const float* Wk = (const float*)d_in[3];
  const float* bk = (const float*)d_in[4];
  const float* Wv = (const float*)d_in[5];
  const float* bv = (const float*)d_in[6];
  const float* Wo = (const float*)d_in[7];
  const float* bo = (const float*)d_in[8];

  float* outp = (float*)d_out;                              // [8,1024,768]
  float* attn = outp + (size_t)BS_ * DM_;                   // [8,12,1024,1024]

  // workspace carve-up (f16 buffers), 256B aligned
  char* ws = (char*)d_ws;
  size_t off = 0;
  auto take = [&](size_t bytes) {
    char* p = ws + off;
    off = (off + bytes + 255) & ~(size_t)255;
    return p;
  };
  const size_t tokBytes = (size_t)BS_ * DM_ * sizeof(_Float16);   // 12.6 MB
  const size_t wBytes   = (size_t)DM_ * DM_ * sizeof(_Float16);   // 1.18 MB
  _Float16* xh  = (_Float16*)take(tokBytes);
  _Float16* wqh = (_Float16*)take(wBytes);
  _Float16* wkh = (_Float16*)take(wBytes);
  _Float16* wvh = (_Float16*)take(wBytes);
  _Float16* woh = (_Float16*)take(wBytes);
  _Float16* qh  = (_Float16*)take(tokBytes);
  _Float16* kh  = (_Float16*)take(tokBytes);
  _Float16* vh  = (_Float16*)take(tokBytes);
  _Float16* mt  = (_Float16*)take((size_t)B_ * H_ * HD_ * HD_ * sizeof(_Float16));
  _Float16* ohb = (_Float16*)take(tokBytes);
  (void)ws_size;

  // 1) converts (8 f32 -> 8 f16 per thread)
  {
    const int nTok8 = (BS_ * DM_) / 8, nW8 = (DM_ * DM_) / 8;
    k_cvt_f16v<<<(nTok8 + 255) / 256, 256, 0, stream>>>(x,  xh,  nTok8);
    k_cvt_f16v<<<(nW8 + 255) / 256, 256, 0, stream>>>(Wq, wqh, nW8);
    k_cvt_f16v<<<(nW8 + 255) / 256, 256, 0, stream>>>(Wk, wkh, nW8);
    k_cvt_f16v<<<(nW8 + 255) / 256, 256, 0, stream>>>(Wv, wvh, nW8);
    k_cvt_f16v<<<(nW8 + 255) / 256, 256, 0, stream>>>(Wo, woh, nW8);
  }

  const int gemmBlocks = (BS_ / 16) * (DM_ / 64) / 8;   // 768
  // 2) q/k/v projections (q carries the 1/sqrt(HD)=1/8 scale)
  k_gemm_xwT<_Float16><<<gemmBlocks, 256, 0, stream>>>(xh, wqh, bq, qh, 0.125f);
  k_gemm_xwT<_Float16><<<gemmBlocks, 256, 0, stream>>>(xh, wkh, bk, kh, 1.0f);
  k_gemm_xwT<_Float16><<<gemmBlocks, 256, 0, stream>>>(xh, wvh, bv, vh, 1.0f);

  const int headBlocks = (B_ * H_ * (S_ / 16)) / 8;     // 768
  // 3) fused energy + softmax -> attention output (single streaming write)
  k_energy_softmax<<<headBlocks, 256, 0, stream>>>(qh, kh, attn);
  // 4) per-head k^T v (64x64, stored transposed), async global->LDS staging
  k_ktv<<<B_ * H_, 128, 0, stream>>>(kh, vh, mt);
  // 5) oh = q @ (k^T v)   (== energy @ v, since no softmax in between)
  k_qm<<<headBlocks, 256, 0, stream>>>(qh, mt, ohb);
  // 6) out = oh @ Wo^T + bo   (f32 out)
  k_gemm_xwT<float><<<gemmBlocks, 256, 0, stream>>>(ohb, woh, bo, outp, 1.0f);
}
